// ScaledDotProductAttention_50843822850339
// MI455X (gfx1250) — matrix operand order repaired and checked by reading, not compiled, not run on hardware
//
#include <hip/hip_runtime.h>
#include <math.h>
#include <stdint.h>

typedef __attribute__((ext_vector_type(16))) _Float16 v16h;
typedef __attribute__((ext_vector_type(8)))  float    v8f;
typedef __attribute__((ext_vector_type(4)))  float    v4f;

#define NH   16
#define SEQ  4096
#define DIM  64
#define QT   16              // query rows per workgroup
#define NKT  (SEQ / 16)      // 256 key tiles of 16
#define NW   8               // waves per workgroup (256 threads, wave32)
#define VROWS 64             // V rows staged per PV step
#define VPAD  65             // padded LDS row stride (floats) -> conflict-free column reads
#define NSTEP (SEQ / VROWS)  // 64 PV steps

// LDS layout (dynamic, 296448 B of the 320 KB WGP LDS):
//   sS    [QT][SEQ]   f32 scores / probabilities        262144 B
//   sRed  [QT][16]    f32 row-reduction slots              1024 B
//   union {
//     sV  [2][VROWS][VPAD] f32 double-buffered V stage    33280 B
//     sPart [NW][QT][DIM]  f32 per-wave PV partials       32768 B
//   }
#define LDS_S_BYTES    (QT * SEQ * 4)
#define LDS_RED_BYTES  (QT * 16 * 4)
#define VSTG_FLOATS    (VROWS * VPAD)
#define LDS_UNION      (2 * VSTG_FLOATS * 4)   // 33280 >= 32768 (sPart)
#define LDS_TOTAL      (LDS_S_BYTES + LDS_RED_BYTES + LDS_UNION)

// Async global->LDS copy via CDNA5 GLOBAL_LOAD_ASYNC_TO_LDS_B128 (ASYNCcnt).
// The clang builtin exists but its params are in LangAS cuda_device/cuda_shared
// (unspellable from HIP), so we issue the instruction with inline asm:
//   VDST = LDS byte address VGPR, VADDR = 64-bit global address pair.
#define USE_ASYNC_LDS 1

__device__ __forceinline__ void async_copy_b128(const float* g, float* l) {
#ifdef USE_ASYNC_LDS
    const uint32_t lds_addr = (uint32_t)(uintptr_t)l;   // LDS aperture: addr[31:0]
    const uint64_t gaddr    = (uint64_t)(uintptr_t)g;
    asm volatile("global_load_async_to_lds_b128 %0, %1, off"
                 :: "v"(lds_addr), "v"(gaddr)
                 : "memory");
#else
    *(v4f*)l = *(const v4f*)g;
#endif
}

__device__ __forceinline__ void wait_async_copies() {
#ifdef USE_ASYNC_LDS
    asm volatile("s_wait_asynccnt 0x0" ::: "memory");
#endif
}

// Stage VROWS x DIM fp32 V rows into LDS with padded stride (dst row = VPAD floats).
// 256 threads x 4 x 16B, fully coalesced.
__device__ __forceinline__ void stage_v(const float* __restrict__ gsrc,
                                        float* __restrict__ ldst, int tid) {
#pragma unroll
    for (int j = 0; j < (VROWS * DIM) / (256 * 4); ++j) {
        const int f   = tid + j * 256;     // float4 index
        const int row = f >> 4;            // 16 float4 per 64-float row
        const int c4  = (f & 15) * 4;
        async_copy_b128(gsrc + row * DIM + c4, ldst + row * VPAD + c4);
    }
}

__global__ __launch_bounds__(256) void fa_fused_attn(
    const float* __restrict__ Q,
    const float* __restrict__ K,
    const float* __restrict__ V,
    float* __restrict__ Out,     // [NH][SEQ][DIM]
    float* __restrict__ Attn)    // [NH][SEQ][SEQ]
{
    extern __shared__ char raw[];
    float* sS    = (float*)(raw);
    float* sRed  = (float*)(raw + LDS_S_BYTES);
    float* sVbuf = (float*)(raw + LDS_S_BYTES + LDS_RED_BYTES);  // double buffer
    float* sPart = sVbuf;                                        // union (used after)

    const int tid  = threadIdx.x;
    const int wave = tid >> 5;
    const int lane = tid & 31;
    const int hs   = lane >> 4;   // half-wave select
    const int ln   = lane & 15;

    const int h     = blockIdx.y;
    const int qbase = blockIdx.x * QT;

    const float* Qh = Q + (size_t)h * SEQ * DIM;
    const float* Kh = K + (size_t)h * SEQ * DIM;
    const float* Vh = V + (size_t)h * SEQ * DIM;

    // ---- Q tile as two 16x32 f16 A-fragments (resident in VGPRs).
    // A layout (16-bit, 16x32): M = lane%16; lanes 0-15 hold K {0..7,16..23},
    // lanes 16-31 hold K {8..15,24..31}.
    v16h aq0, aq1;
    {
        const float* qr = Qh + (size_t)(qbase + ln) * DIM;
#pragma unroll
        for (int i = 0; i < 8; ++i) {
            aq0[i]     = (_Float16)qr[ 0 + hs * 8 + i];
            aq0[i + 8] = (_Float16)qr[16 + hs * 8 + i];
            aq1[i]     = (_Float16)qr[32 + hs * 8 + i];
            aq1[i + 8] = (_Float16)qr[48 + hs * 8 + i];
        }
    }

    // ================= Phase 1: S = Q K^T / sqrt(D) into LDS =================
    // B layout (16-bit, 32x16): N = lane%16; lanes 0-15 hold K {0..7,16..23},
    // lanes 16-31 K {8..15,24..31}.  B's K-dim is the head dim (contiguous fp32).
    for (int j = wave; j < NKT; j += NW) {
        const float* kr = Kh + (size_t)(j * 16 + ln) * DIM;
        v16h bk0, bk1;
#pragma unroll
        for (int i = 0; i < 8; ++i) {
            bk0[i]     = (_Float16)kr[ 0 + hs * 8 + i];
            bk0[i + 8] = (_Float16)kr[16 + hs * 8 + i];
            bk1[i]     = (_Float16)kr[32 + hs * 8 + i];
            bk1[i + 8] = (_Float16)kr[48 + hs * 8 + i];
        }
        v8f c = {};
        c = __builtin_amdgcn_wmma_f32_16x16x32_f16(false, aq0, false, bk0, (short)0, c, false, false);
        c = __builtin_amdgcn_wmma_f32_16x16x32_f16(false, aq1, false, bk1, (short)0, c, false, false);
#pragma unroll
        for (int r = 0; r < 8; ++r) {
            const int m = r + hs * 8;                     // C: VGPR r -> M = r + 8*hs, N = lane%16
            sS[m * SEQ + j * 16 + ln] = c[r] * 0.125f;    // 1/sqrt(64)
        }
    }
    __syncthreads();

    // ================= Phase 2: row softmax (float4 everywhere) ==============
    // 16 threads per row; thread owns float4 chunks  cc*4 + 64*j.
    const int row = tid >> 4;
    const int cc  = tid & 15;
    float* srow = sS + row * SEQ;
    v4f* srow4 = (v4f*)srow;

    float lmax = -3.4e38f;
    for (int j = 0; j < SEQ / 64; ++j) {
        v4f x = srow4[j * 16 + cc];
        lmax = fmaxf(fmaxf(fmaxf(lmax, x[0]), fmaxf(x[1], x[2])), x[3]);
    }
    sRed[row * 16 + cc] = lmax;
    __syncthreads();
    float rmax = -3.4e38f;
#pragma unroll
    for (int i = 0; i < 16; ++i) rmax = fmaxf(rmax, sRed[row * 16 + i]);
    __syncthreads();

    float lsum = 0.f;
    for (int j = 0; j < SEQ / 64; ++j) {
        v4f x = srow4[j * 16 + cc];
        v4f e;
        e[0] = __expf(x[0] - rmax);
        e[1] = __expf(x[1] - rmax);
        e[2] = __expf(x[2] - rmax);
        e[3] = __expf(x[3] - rmax);
        srow4[j * 16 + cc] = e;
        lsum += e[0] + e[1] + e[2] + e[3];
    }
    sRed[row * 16 + cc] = lsum;
    __syncthreads();
    float rsum = 0.f;
#pragma unroll
    for (int i = 0; i < 16; ++i) rsum += sRed[row * 16 + i];
    const float rinv = 1.0f / rsum;

    // Normalize in LDS and stream P to global attn: b128 stores, 256 B
    // contiguous per 16-lane group -- this is the mandatory 1 GB HBM write.
    v4f* attn_row4 = (v4f*)(Attn + ((size_t)h * SEQ + qbase + row) * SEQ);
    for (int j = 0; j < SEQ / 64; ++j) {
        v4f p = srow4[j * 16 + cc];
        p[0] *= rinv; p[1] *= rinv; p[2] *= rinv; p[3] *= rinv;
        srow4[j * 16 + cc]   = p;
        attn_row4[j * 16 + cc] = p;
    }
    __syncthreads();

    // ================= Phase 3: O = P V, lock-step staged V chunks ===========
    // wave -> (K-half wk = wave>>2, N-tile w4 = wave&3); 1 WMMA per wave/step.
    const int w4 = wave & 3;
    const int wk = wave >> 2;
    float* sV0 = sVbuf;
    float* sV1 = sVbuf + VSTG_FLOATS;

    stage_v(Vh, sV0, tid);   // step 0
    v8f acc = {};
    for (int step = 0; step < NSTEP; ++step) {
        wait_async_copies();
        __syncthreads();                      // staged chunk for `step` visible
        if (step + 1 < NSTEP)
            stage_v(Vh + (size_t)(step + 1) * VROWS * DIM,
                    ((step + 1) & 1) ? sV1 : sV0, tid);

        const float* cur = (step & 1) ? sV1 : sV0;
        const int kbw = step * VROWS + wk * 32;

        // A-fragment: P tile 16x32 from LDS (f32 -> f16), contiguous b128 reads.
        v16h ap;
#pragma unroll
        for (int i = 0; i < 8; ++i) {
            ap[i]     = (_Float16)sS[ln * SEQ + kbw +      hs * 8 + i];
            ap[i + 8] = (_Float16)sS[ln * SEQ + kbw + 16 + hs * 8 + i];
        }
        // B-fragment: staged V, column reads spread over 16 banks (VPAD=65).
        const float* vcol = cur + (wk * 32 + hs * 8) * VPAD + w4 * 16 + ln;
        v16h bv;
#pragma unroll
        for (int i = 0; i < 8; ++i) {
            bv[i]     = (_Float16)vcol[i * VPAD];
            bv[i + 8] = (_Float16)vcol[(i + 16) * VPAD];
        }

        acc = __builtin_amdgcn_wmma_f32_16x16x32_f16(false, ap, false, bv, (short)0, acc, false, false);
    }
    __syncthreads();   // all sV reads done before the union buffer is reused

    // Spill per-wave partials and reduce the two K-halves per N-tile.
#pragma unroll
    for (int r = 0; r < 8; ++r) {
        const int m = r + hs * 8;
        sPart[((size_t)wave * QT + m) * DIM + w4 * 16 + ln] = acc[r];
    }
    __syncthreads();

    for (int idx = tid; idx < QT * DIM; idx += 256) {
        const int m = idx >> 6;
        const int d = idx & 63;
        const int t = d >> 4;                 // owning N-tile
        const float s = sPart[((size_t)t * QT + m) * DIM + d]
                      + sPart[((size_t)(t + 4) * QT + m) * DIM + d];
        Out[((size_t)h * SEQ + qbase + m) * DIM + d] = s;
    }
}

extern "C" void kernel_launch(void* const* d_in, const int* in_sizes, int n_in,
                              void* d_out, int out_size, void* d_ws, size_t ws_size,
                              hipStream_t stream) {
    (void)in_sizes; (void)n_in; (void)d_ws; (void)ws_size; (void)out_size;
    const float* q = (const float*)d_in[0];
    const float* k = (const float*)d_in[1];
    const float* v = (const float*)d_in[2];
    float* out  = (float*)d_out;                       // [16,4096,64]
    float* attn = out + (size_t)NH * SEQ * DIM;        // [16,4096,4096]

    (void)hipFuncSetAttribute((const void*)fa_fused_attn,
                              hipFuncAttributeMaxDynamicSharedMemorySize, LDS_TOTAL);

    dim3 grid(SEQ / QT, NH);   // 256 query tiles x 16 heads
    dim3 block(256);           // 8 waves (wave32)
    fa_fused_attn<<<grid, block, LDS_TOTAL, stream>>>(q, k, v, out, attn);
}
